// ViLT_54511724921446
// MI455X (gfx1250) — matrix operand order repaired and checked by reading, not compiled, not run on hardware
//
#include <hip/hip_runtime.h>
#include <hip/hip_bf16.h>

typedef __attribute__((ext_vector_type(16))) _Float16 v16h;
typedef __attribute__((ext_vector_type(8)))  _Float16 v8h;
typedef __attribute__((ext_vector_type(4)))  _Float16 v4h;
typedef __attribute__((ext_vector_type(8)))  float    v8f;
typedef __attribute__((ext_vector_type(4)))  int      v4i;

#if defined(__HIP_DEVICE_COMPILE__) && defined(__has_builtin)
#if __has_builtin(__builtin_amdgcn_global_load_async_to_lds_b128) && \
    __has_builtin(__builtin_amdgcn_s_wait_asynccnt)
#define HAVE_ASYNC_LDS 1
#endif
#endif
#ifndef HAVE_ASYNC_LDS
#define HAVE_ASYNC_LDS 0
#endif

#define ASG __attribute__((address_space(1)))
#define ASL __attribute__((address_space(3)))

// ---- problem dims ----
static constexpr int LYR   = 12;
static constexpr int CD    = 768;
static constexpr int FF    = 3072;
static constexpr int NH    = 12;
static constexpr int DHD   = 64;
static constexpr int BSZ   = 64;
static constexpr int PP    = 196;
static constexpr int TT    = 40;
static constexpr int NTOK  = 237;      // TT + 1 + PP
static constexpr int NPAD  = 240;
static constexpr int MROWS = BSZ * NTOK;   // 15168
static constexpr int MPATCH = BSZ * PP;    // 12544
static constexpr int QKVD  = 3 * CD;       // 2304

__device__ __forceinline__ v8f zero8() { v8f z = {0.f,0.f,0.f,0.f,0.f,0.f,0.f,0.f}; return z; }

// ---------------- f32 -> f16 weight conversion ----------------
__global__ void cvt_f16_kernel(const float* __restrict__ src, _Float16* __restrict__ dst, int n4) {
  for (int i = blockIdx.x*blockDim.x + threadIdx.x; i < n4; i += gridDim.x*blockDim.x) {
    float4 v = ((const float4*)src)[i];
    v4h o;
    o[0] = (_Float16)v.x; o[1] = (_Float16)v.y; o[2] = (_Float16)v.z; o[3] = (_Float16)v.w;
    *(v4h*)(dst + (size_t)i*4) = o;
  }
}

// ---------------- LayerNorm helper (one row per 256-thread block) ----------------
template <typename OutT>
__device__ __forceinline__ void row_ln_store(float v0, float v1, float v2,
                                             const float* __restrict__ w,
                                             const float* __restrict__ b,
                                             float eps, OutT* __restrict__ out, int tid) {
  __shared__ float s1[256], s2[256];
  s1[tid] = v0 + v1 + v2;
  s2[tid] = v0*v0 + v1*v1 + v2*v2;
  __syncthreads();
  for (int off = 128; off > 0; off >>= 1) {
    if (tid < off) { s1[tid] += s1[tid+off]; s2[tid] += s2[tid+off]; }
    __syncthreads();
  }
  float mean = s1[0] * (1.0f/768.0f);
  float var  = s2[0] * (1.0f/768.0f) - mean*mean;
  float r = rsqrtf(var + eps);
  out[tid]     = (OutT)((v0-mean)*r*w[tid]     + b[tid]);
  out[tid+256] = (OutT)((v1-mean)*r*w[tid+256] + b[tid+256]);
  out[tid+512] = (OutT)((v2-mean)*r*w[tid+512] + b[tid+512]);
}

__global__ void ln_rows_kernel(const float* __restrict__ x, const float* __restrict__ w,
                               const float* __restrict__ b, _Float16* __restrict__ out,
                               float eps) {
  size_t row = blockIdx.x;
  int tid = threadIdx.x;
  const float* xr = x + row * CD;
  row_ln_store<_Float16>(xr[tid], xr[tid+256], xr[tid+512], w, b, eps, out + row*CD, tid);
}

__global__ void embed_text_kernel(const int* __restrict__ txt_in,
                                  const float* __restrict__ w_embed,
                                  const float* __restrict__ t_pos,
                                  const float* __restrict__ t_mod,
                                  const float* __restrict__ w, const float* __restrict__ b,
                                  float* __restrict__ xout) {
  int row = blockIdx.x;                  // 0..B*T-1
  int bb = row / TT, t = row % TT;
  int tid = threadIdx.x;
  int tok = txt_in[row];
  float v[3];
#pragma unroll
  for (int i = 0; i < 3; ++i) {
    int c = tid + i*256;
    v[i] = w_embed[(size_t)tok*CD + c] + t_pos[(size_t)t*CD + c] + t_mod[c];
  }
  float* out = xout + ((size_t)(bb*NTOK + t)) * CD;
  row_ln_store<float>(v[0], v[1], v[2], w, b, 1e-5f, out, tid);
}

__global__ void embed_vis_kernel(const float* __restrict__ vtmp,
                                 const float* __restrict__ v_cls,
                                 const float* __restrict__ v_pos,
                                 const float* __restrict__ v_mod,
                                 const float* __restrict__ w, const float* __restrict__ b,
                                 float* __restrict__ xout) {
  int row = blockIdx.x;                  // 0..B*(P+1)-1
  int bb = row / (PP+1), p = row % (PP+1);
  int tid = threadIdx.x;
  float v[3];
#pragma unroll
  for (int i = 0; i < 3; ++i) {
    int c = tid + i*256;
    float e = (p == 0) ? v_cls[c] : vtmp[((size_t)(bb*PP + p - 1))*CD + c];
    v[i] = e + v_pos[(size_t)p*CD + c] + v_mod[c];
  }
  float* out = xout + ((size_t)(bb*NTOK + TT + p)) * CD;
  row_ln_store<float>(v[0], v[1], v[2], w, b, 1e-5f, out, tid);
}

__global__ void build_mask_kernel(const int* __restrict__ txt_mask,
                                  const int* __restrict__ vis_mask,
                                  int* __restrict__ mask) {
  for (int idx = blockIdx.x*blockDim.x + threadIdx.x; idx < BSZ*NPAD;
       idx += gridDim.x*blockDim.x) {
    int bb = idx / NPAD, t = idx % NPAD;
    int m;
    if (t < TT)           m = txt_mask[bb*TT + t];
    else if (t == TT)     m = 1;
    else if (t < NTOK)    m = vis_mask[bb*PP + (t - TT - 1)];
    else                  m = 0;
    mask[idx] = m;
  }
}

// ---------------- Generic WMMA GEMM:  out = A[M,K] * W[N,K]^T (+bias, epilogue) ----
// Double-buffered LDS pipeline; B tile staged via async global->LDS when available.
// EPI: 0 = store f32, 1 = f32 residual add, 2 = store f16, 3 = GELU -> f16
// sA rows are stored contiguously: p -> k, so the A fragment for lane-half lh is
// the contiguous 16 halves at [m][lh*16] (k = lh*16 + s), matching sB.
template <bool AF32, int EPI>
__global__ __launch_bounds__(256)
void gemm_wmma_kernel(const void* __restrict__ Aptr, const _Float16* __restrict__ W,
                      const float* __restrict__ bias,
                      float* __restrict__ outF, _Float16* __restrict__ outH,
                      int M, int N, int K) {
  __shared__ __attribute__((aligned(64))) _Float16 sA[2][128][48];
  __shared__ __attribute__((aligned(64))) _Float16 sB[2][128][48];
  int tid  = threadIdx.x;
  int lane = tid & 31, wave = tid >> 5;
  int ll = lane & 15, lh = lane >> 4;
  int bm = blockIdx.y * 128, bn = blockIdx.x * 128;
  int wr = (wave & 3) * 32;    // wave row offset (2 frags)
  int wc = (wave >> 2) * 64;   // wave col offset (4 frags)

  v8f acc[2][4];
#pragma unroll
  for (int i = 0; i < 2; ++i)
#pragma unroll
    for (int j = 0; j < 4; ++j) acc[i][j] = zero8();

  int ldrow = tid >> 1;
  int ldk   = (tid & 1) * 16;
  int dlo   = ldk;              // contiguous dest: 0 or 16
  const _Float16* A16 = (const _Float16*)Aptr;
  const float*    A32 = (const float*)Aptr;
  int gmL = bm + ldrow; if (gmL >= M) gmL = M - 1;
  int gnL = bn + ldrow; if (gnL >= N) gnL = N - 1;
  const int nk = K >> 5;

  // --- staging helpers ---
  auto loadA = [&](int kt, v8h& lo, v8h& hi) {
    if (AF32) {
      const float4* src = (const float4*)(A32 + (size_t)gmL*K + kt*32 + ldk);
#pragma unroll
      for (int i = 0; i < 2; ++i) {
        float4 a = src[i], b = src[i+2];
        lo[4*i+0]=(_Float16)a.x; lo[4*i+1]=(_Float16)a.y; lo[4*i+2]=(_Float16)a.z; lo[4*i+3]=(_Float16)a.w;
        hi[4*i+0]=(_Float16)b.x; hi[4*i+1]=(_Float16)b.y; hi[4*i+2]=(_Float16)b.z; hi[4*i+3]=(_Float16)b.w;
      }
    } else {
      v16h s = *(const v16h*)(A16 + (size_t)gmL*K + kt*32 + ldk);
      lo = __builtin_shufflevector(s, s, 0,1,2,3,4,5,6,7);
      hi = __builtin_shufflevector(s, s, 8,9,10,11,12,13,14,15);
    }
  };
  auto storeA = [&](int buf, v8h lo, v8h hi) {
    *(v8h*)&sA[buf][ldrow][dlo]      = lo;
    *(v8h*)&sA[buf][ldrow][dlo + 8] = hi;
  };
#if HAVE_ASYNC_LDS
  auto issueB = [&](int kt, int buf) {
    const char* g = (const char*)(W + (size_t)gnL*K + kt*32 + ldk);
    char* l = (char*)&sB[buf][ldrow][ldk];
    __builtin_amdgcn_global_load_async_to_lds_b128((ASG v4i*)g,      (ASL v4i*)l,      0, 0);
    __builtin_amdgcn_global_load_async_to_lds_b128((ASG v4i*)(g+16), (ASL v4i*)(l+16), 0, 0);
  };
#endif

  // --- prologue: fill buffer 0 ---
  {
    v8h alo, ahi;
    loadA(0, alo, ahi);
#if HAVE_ASYNC_LDS
    issueB(0, 0);
#else
    *(v16h*)&sB[0][ldrow][ldk] = *(const v16h*)(W + (size_t)gnL*K + ldk);
#endif
    storeA(0, alo, ahi);
#if HAVE_ASYNC_LDS
    __builtin_amdgcn_s_wait_asynccnt(0);
#endif
    __syncthreads();
  }

  for (int kt = 0; kt < nk; ++kt) {
    int cur = kt & 1, nxt = cur ^ 1;
    bool more = (kt + 1 < nk);
    v8h nalo, nahi;
#if !HAVE_ASYNC_LDS
    v16h nb;
#endif
    if (more) {
      loadA(kt + 1, nalo, nahi);
#if HAVE_ASYNC_LDS
      issueB(kt + 1, nxt);
#else
      nb = *(const v16h*)(W + (size_t)gnL*K + (kt+1)*32 + ldk);
#endif
    }
    // compute on current buffer
    v16h af[2];
#pragma unroll
    for (int mi = 0; mi < 2; ++mi)
      af[mi] = *(const v16h*)&sA[cur][wr + mi*16 + ll][lh * 16];
#pragma unroll
    for (int ni = 0; ni < 4; ++ni) {
      v16h bf = *(const v16h*)&sB[cur][wc + ni*16 + ll][lh * 16];
#pragma unroll
      for (int mi = 0; mi < 2; ++mi)
        acc[mi][ni] = __builtin_amdgcn_wmma_f32_16x16x32_f16(
            false, af[mi], false, bf, (short)0, acc[mi][ni], false, false);
    }
    if (more) {
      storeA(nxt, nalo, nahi);
#if !HAVE_ASYNC_LDS
      *(v16h*)&sB[nxt][ldrow][ldk] = nb;
#endif
    }
#if HAVE_ASYNC_LDS
    __builtin_amdgcn_s_wait_asynccnt(0);
#endif
    __syncthreads();
  }

  // epilogue (D layout: row = r + 8*lh, col = ll)
#pragma unroll
  for (int mi = 0; mi < 2; ++mi) {
#pragma unroll
    for (int ni = 0; ni < 4; ++ni) {
      int gn = bn + wc + ni*16 + ll;
      float bv = bias ? bias[gn] : 0.0f;
#pragma unroll
      for (int r = 0; r < 8; ++r) {
        int gm = bm + wr + mi*16 + r + lh*8;
        if (gm < M) {
          float v = acc[mi][ni][r] + bv;
          if (EPI == 0)      outF[(size_t)gm*N + gn] = v;
          else if (EPI == 1) outF[(size_t)gm*N + gn] += v;
          else if (EPI == 2) outH[(size_t)gm*N + gn] = (_Float16)v;
          else {
            float g = 0.5f * v * (1.0f + erff(v * 0.70710678118f));
            outH[(size_t)gm*N + gn] = (_Float16)g;
          }
        }
      }
    }
  }
}

// ---------------- Attention: one wave per (b, head, 16-row q tile) --------------
__global__ __launch_bounds__(32)
void attn_kernel(const _Float16* __restrict__ qkv, const int* __restrict__ mask,
                 _Float16* __restrict__ o16) {
  __shared__ __attribute__((aligned(64))) _Float16 sP[16][256];
  int id = blockIdx.x;
  int qt = id % 15; id /= 15;
  int h  = id % NH;
  int b  = id / NH;
  int lane = threadIdx.x;
  int ll = lane & 15, lh = lane >> 4;
  size_t base = (size_t)b * NTOK * QKVD;
  const int* mrow = mask + b * NPAD;

  // Q fragments (A layout), dh = 0..63 -> two 16x32 frags
  int qtok = qt*16 + ll; if (qtok >= NTOK) qtok = NTOK - 1;
  const _Float16* qp = qkv + base + (size_t)qtok*QKVD + h*DHD;
  v16h qa[2];
#pragma unroll
  for (int f = 0; f < 2; ++f) {
    int k0 = f*32 + lh*16;
    v8h lo = *(const v8h*)(qp + k0);
    v8h hi = *(const v8h*)(qp + k0 + 8);
    v16h a;
#pragma unroll
    for (int i = 0; i < 8; ++i) { a[i] = lo[i]; a[8+i] = hi[i]; }
    qa[f] = a;
  }

  // hoisted row-mask flags (unconditional clamped loads, no branches)
  bool rowMasked[8];
#pragma unroll
  for (int r = 0; r < 8; ++r) {
    int rtok = qt*16 + r + lh*8;
    int rc = (rtok < NTOK) ? rtok : NTOK - 1;
    int mr = mrow[rc];
    rowMasked[r] = (rtok < NTOK) && (mr == 0);
  }

  v8f S[15];
#pragma unroll
  for (int j = 0; j < 15; ++j) {
    int ktok = j*16 + ll;
    int ktc  = (ktok < NTOK) ? ktok : NTOK - 1;
    const _Float16* kp = qkv + base + (size_t)ktc*QKVD + CD + h*DHD + lh*16;
    v16h b0 = *(const v16h*)(kp);        // K frag (B layout): k = lh*16+i
    v16h b1 = *(const v16h*)(kp + 32);   // k = 32 + lh*16+i
    v8f a = zero8();
    a = __builtin_amdgcn_wmma_f32_16x16x32_f16(false, qa[0], false, b0, (short)0, a, false, false);
    a = __builtin_amdgcn_wmma_f32_16x16x32_f16(false, qa[1], false, b1, (short)0, a, false, false);
    bool pad = (ktok >= NTOK);
    bool colMasked = (mrow[ktc] == 0);
#pragma unroll
    for (int r = 0; r < 8; ++r) {
      float s = a[r] * 0.125f;                        // 1/sqrt(64)
      s = (colMasked || rowMasked[r]) ? -1.0e30f : s; // reference mask value
      s = pad ? -3.0e38f : s;                         // padding strictly excluded
      a[r] = s;
    }
    S[j] = a;
  }

  // row softmax (rows live on 16-lane halves; reduce with xor shuffles <16)
  float rmax[8], rsum[8];
#pragma unroll
  for (int r = 0; r < 8; ++r) rmax[r] = -3.4e38f;
#pragma unroll
  for (int j = 0; j < 15; ++j)
#pragma unroll
    for (int r = 0; r < 8; ++r) rmax[r] = fmaxf(rmax[r], S[j][r]);
#pragma unroll
  for (int off = 1; off < 16; off <<= 1)
#pragma unroll
    for (int r = 0; r < 8; ++r) rmax[r] = fmaxf(rmax[r], __shfl_xor(rmax[r], off, 32));
#pragma unroll
  for (int r = 0; r < 8; ++r) rsum[r] = 0.0f;
#pragma unroll
  for (int j = 0; j < 15; ++j)
#pragma unroll
    for (int r = 0; r < 8; ++r) {
      float e = __expf(S[j][r] - rmax[r]);
      S[j][r] = e;
      rsum[r] += e;
    }
#pragma unroll
  for (int off = 1; off < 16; off <<= 1)
#pragma unroll
    for (int r = 0; r < 8; ++r) rsum[r] += __shfl_xor(rsum[r], off, 32);
  float rinv[8];
#pragma unroll
  for (int r = 0; r < 8; ++r) rinv[r] = 1.0f / rsum[r];

  // write P (f16) to LDS in plain [row][col] order, zero the 240..255 pad
#pragma unroll
  for (int j = 0; j < 15; ++j)
#pragma unroll
    for (int r = 0; r < 8; ++r)
      sP[r + lh*8][j*16 + ll] = (_Float16)(S[j][r] * rinv[r]);
  for (int idx = lane; idx < 256; idx += 32)
    sP[idx >> 4][240 + (idx & 15)] = (_Float16)0.0f;
  __syncthreads();

  // O[16,64] = P[16,240] * V[240,64]; contraction over tokens in chunks of 32
#pragma unroll
  for (int c = 0; c < 4; ++c) {
    v8f a = zero8();
#pragma unroll
    for (int jj = 0; jj < 8; ++jj) {
      int t0 = jj*32 + lh*8;
      v8h lo = *(const v8h*)&sP[ll][t0];
      v8h hi = *(const v8h*)&sP[ll][t0 + 16];
      v16h pa;
#pragma unroll
      for (int i = 0; i < 8; ++i) { pa[i] = lo[i]; pa[8+i] = hi[i]; }
      v16h vb;
      int dh = c*16 + ll;
#pragma unroll
      for (int i = 0; i < 16; ++i) {
        int t = jj*32 + lh*8 + i + (i >> 3)*8;
        int tc = (t < NTOK) ? t : NTOK - 1;
        vb[i] = qkv[base + (size_t)tc*QKVD + 2*CD + h*DHD + dh];
      }
      a = __builtin_amdgcn_wmma_f32_16x16x32_f16(false, pa, false, vb, (short)0, a, false, false);
    }
#pragma unroll
    for (int r = 0; r < 8; ++r) {
      int rtok = qt*16 + r + lh*8;
      if (rtok < NTOK)
        o16[((size_t)(b*NTOK + rtok))*CD + h*DHD + c*16 + ll] = (_Float16)a[r];
    }
  }
}

// ---------------- host orchestration ----------------
extern "C" void kernel_launch(void* const* d_in, const int* in_sizes, int n_in,
                              void* d_out, int out_size, void* d_ws, size_t ws_size,
                              hipStream_t stream) {
  (void)in_sizes; (void)n_in; (void)out_size; (void)ws_size;
  const float* vis_in   = (const float*)d_in[0];
  const int*   txt_in   = (const int*)d_in[1];
  const int*   vis_mask = (const int*)d_in[2];
  const int*   txt_mask = (const int*)d_in[3];
  const float* patch_w  = (const float*)d_in[4];
  const float* patch_b  = (const float*)d_in[5];
  const float* w_embed  = (const float*)d_in[6];
  const float* t_pos    = (const float*)d_in[7];
  const float* v_pos    = (const float*)d_in[8];
  const float* v_cls    = (const float*)d_in[9];
  const float* v_mod    = (const float*)d_in[10];
  const float* t_mod    = (const float*)d_in[11];
  const float* vn_w     = (const float*)d_in[12];
  const float* vn_b     = (const float*)d_in[13];
  const float* tn_w     = (const float*)d_in[14];
  const float* tn_b     = (const float*)d_in[15];
  const float* qkv_w    = (const float*)d_in[16];
  const float* qkv_b    = (const float*)d_in[17];
  const float* proj_w   = (const float*)d_in[18];
  const float* proj_b   = (const float*)d_in[19];
  const float* n1_w     = (const float*)d_in[20];
  const float* n1_b     = (const float*)d_in[21];
  const float* n2_w     = (const float*)d_in[22];
  const float* n2_b     = (const float*)d_in[23];
  const float* fc1_w    = (const float*)d_in[24];
  const float* fc1_b    = (const float*)d_in[25];
  const float* fc2_w    = (const float*)d_in[26];
  const float* fc2_b    = (const float*)d_in[27];

  float* x = (float*)d_out;       // residual stream [M, C] f32
  char* ws = (char*)d_ws;
  const size_t h16_bytes  = (size_t)MROWS * CD * 2;      // 23,298,048
  const size_t qkv_bytes  = (size_t)MROWS * QKVD * 2;    // 69,894,144
  const size_t o16_bytes  = (size_t)MROWS * CD * 2;      // 23,298,048
  const size_t mask_bytes = (size_t)BSZ * NPAD * 4;      // 61,440
  _Float16* h16   = (_Float16*)ws;
  _Float16* qkv16 = (_Float16*)(ws + h16_bytes);
  _Float16* o16   = (_Float16*)(ws + h16_bytes + qkv_bytes);
  _Float16* g16   = qkv16;                 // fc1 output aliases (qkv16+o16) region
  float*    vtmp  = (float*)qkv16;         // patch temp aliases qkv16 region
  int*      mask  = (int*)(ws + h16_bytes + qkv_bytes + o16_bytes);
  _Float16* wbuf  = (_Float16*)(ws + h16_bytes + qkv_bytes + o16_bytes + mask_bytes);
  // wbuf: 4,718,592 B max (fc1 weights f16), reused sequentially (stream-ordered)

  const int MT = (MROWS + 127) / 128;      // 119

  // ---- embeddings ----
  build_mask_kernel<<<64, 256, 0, stream>>>(txt_mask, vis_mask, mask);
  cvt_f16_kernel<<<1024, 256, 0, stream>>>(patch_w, wbuf, CD*CD/4);
  gemm_wmma_kernel<true, 0><<<dim3(CD/128, MPATCH/128), 256, 0, stream>>>(
      vis_in, wbuf, patch_b, vtmp, (_Float16*)nullptr, MPATCH, CD, CD);
  embed_text_kernel<<<BSZ*TT, 256, 0, stream>>>(txt_in, w_embed, t_pos, t_mod, tn_w, tn_b, x);
  embed_vis_kernel<<<BSZ*(PP+1), 256, 0, stream>>>(vtmp, v_cls, v_pos, v_mod, vn_w, vn_b, x);

  // ---- transformer layers ----
  for (int l = 0; l < LYR; ++l) {
    ln_rows_kernel<<<MROWS, 256, 0, stream>>>(x, n1_w + l*CD, n1_b + l*CD, h16, 1e-6f);
    cvt_f16_kernel<<<1024, 256, 0, stream>>>(qkv_w + (size_t)l*QKVD*CD, wbuf, QKVD*CD/4);
    gemm_wmma_kernel<false, 2><<<dim3(QKVD/128, MT), 256, 0, stream>>>(
        h16, wbuf, qkv_b + (size_t)l*QKVD, (float*)nullptr, qkv16, MROWS, QKVD, CD);
    attn_kernel<<<BSZ*NH*15, 32, 0, stream>>>(qkv16, mask, o16);
    cvt_f16_kernel<<<1024, 256, 0, stream>>>(proj_w + (size_t)l*CD*CD, wbuf, CD*CD/4);
    gemm_wmma_kernel<false, 1><<<dim3(CD/128, MT), 256, 0, stream>>>(
        o16, wbuf, proj_b + (size_t)l*CD, x, (_Float16*)nullptr, MROWS, CD, CD);
    ln_rows_kernel<<<MROWS, 256, 0, stream>>>(x, n2_w + l*CD, n2_b + l*CD, h16, 1e-6f);
    cvt_f16_kernel<<<1024, 256, 0, stream>>>(fc1_w + (size_t)l*FF*CD, wbuf, FF*CD/4);
    gemm_wmma_kernel<false, 3><<<dim3(FF/128, MT), 256, 0, stream>>>(
        h16, wbuf, fc1_b + (size_t)l*FF, (float*)nullptr, g16, MROWS, FF, CD);
    cvt_f16_kernel<<<1024, 256, 0, stream>>>(fc2_w + (size_t)l*CD*FF, wbuf, CD*FF/4);
    gemm_wmma_kernel<false, 1><<<dim3(CD/128, MT), 256, 0, stream>>>(
        g16, wbuf, fc2_b + (size_t)l*CD, x, (_Float16*)nullptr, MROWS, CD, FF);
  }
}
